// Head_Motion_Prediction_90872918049532
// MI455X (gfx1250) — matrix operand order repaired and checked by reading, not compiled, not run on hardware
//
#include <hip/hip_runtime.h>

// CDNA5 / gfx1250 — wave32, WMMA f32_16x16x32_f16.
// Layout trick: gates on M, batch on N  =>  accumulator (lane=batch, vgpr=unit)
// is lane-aligned with the B-matrix input layout (lane=batch, elem=K), so the
// LSTM recurrence needs only per-lane f32->f16 packing + a few SWAPX16 swizzles.

typedef __attribute__((ext_vector_type(16))) _Float16 v16h;
typedef __attribute__((ext_vector_type(8)))  float    v8f;

#define WPB 4          // waves per block (128 threads)
#define SEQ 18
#define HID 10

__device__ __forceinline__ float lane_xor16(float v) {
  // ds_swizzle SWAPX16: offset = xor(0x10)<<10 | and(0x1f) = 0x401F
  return __int_as_float(__builtin_amdgcn_ds_swizzle(__float_as_int(v), 0x401F));
}
__device__ __forceinline__ float sig_(float x) {
  return __builtin_amdgcn_rcpf(1.0f + __expf(-x));
}
__device__ __forceinline__ float tanh_(float x) {
  float e = __expf(-2.0f * x);
  return (1.0f - e) * __builtin_amdgcn_rcpf(1.0f + e);
}

#define WMMA_F16(Am, Bm, Cm) \
  __builtin_amdgcn_wmma_f32_16x16x32_f16(false, (Am), false, (Bm), (short)0, (Cm), false, false)

__global__ __launch_bounds__(WPB * 32)
void hmp_lstm_wmma_kernel(
    const float* __restrict__ x,
    const float* __restrict__ conv_w, const float* __restrict__ conv_b,
    const float* __restrict__ bn_g,   const float* __restrict__ bn_b,
    const float* __restrict__ bn_m,   const float* __restrict__ bn_v,
    const float* __restrict__ Wih0,   const float* __restrict__ Whh0,
    const float* __restrict__ bih0,   const float* __restrict__ bhh0,
    const float* __restrict__ Wih1,   const float* __restrict__ Whh1,
    const float* __restrict__ bih1,   const float* __restrict__ bhh1,
    const float* __restrict__ fc1w,   const float* __restrict__ fc1b,
    const float* __restrict__ fc2w,   const float* __restrict__ fc2b,
    float* __restrict__ out, int Btot)
{
  __shared__ uint2 sFeat[WPB * 16 * SEQ];   // [wave][b][t] : 3 f16 feats + pad

  const int lane = threadIdx.x & 31;
  const int wave = threadIdx.x >> 5;
  const int b16  = lane & 15;               // batch column within this wave's tile
  const int hi   = lane >> 4;               // lane half: selects K/M sub-range
  const bool lo  = (hi == 0);
  const int bG0  = (blockIdx.x * WPB + wave) * 16 + b16;
  const int bG   = (bG0 < Btot) ? bG0 : (Btot - 1);   // clamp for loads
  const int sBase = wave * 16 * SEQ;

  // ---------------- Phase 1: conv1d(k=3) + BN(eval) + ReLU -> LDS ------------
  {
    const int tBase = hi * 9;               // lanes 0-15: t=0..8, lanes 16-31: t=9..17
    float scl[3], off[3], t0[3], t1[3], t2[3];
#pragma unroll
    for (int c = 0; c < 3; ++c) {
      float sc = bn_g[c] / sqrtf(bn_v[c] + 1e-5f);
      scl[c] = sc;
      off[c] = (conv_b[c] - bn_m[c]) * sc + bn_b[c];
      t0[c] = conv_w[c*3+0]; t1[c] = conv_w[c*3+1]; t2[c] = conv_w[c*3+2];
    }
    float xr[3][11];
#pragma unroll
    for (int c = 0; c < 3; ++c)
#pragma unroll
      for (int k = 0; k < 11; ++k)
        xr[c][k] = x[(size_t)bG * 60 + c * 20 + tBase + k];
#pragma unroll
    for (int t9 = 0; t9 < 9; ++t9) {
      union { uint2 u; _Float16 h[4]; } q;
#pragma unroll
      for (int c = 0; c < 3; ++c) {
        float s = xr[c][t9]*t0[c] + xr[c][t9+1]*t1[c] + xr[c][t9+2]*t2[c];
        float y = fmaxf(s * scl[c] + off[c], 0.0f);
        q.h[c] = (_Float16)y;
      }
      q.h[3] = (_Float16)0.0f;
      sFeat[sBase + b16 * SEQ + tBase + t9] = q.u;
    }
  }
  __syncthreads();

  // ---------------- Phase 2: build constant A tiles (weights, f16) -----------
  // A is 16x32 (M=gate-unit, K). k order used here for A and B: lane L (M=L%16, h=L/16),
// element j  ->  K = 16*h + j.
  const int ml = b16;   // M row this lane supplies
  v16h A0[4], A1[4];
#pragma unroll
  for (int g = 0; g < 4; ++g) {             // gate order i,f,g,o
    v16h a0, a1;
#pragma unroll
    for (int j = 0; j < 16; ++j) {
      const int K = 16 * hi + j;
      float v0 = 0.0f, v1 = 0.0f;
      if (ml < HID) {
        const int row = g * HID + ml;
        // layer0: K0..9 = W_hh0 (h0), K16..18 = W_ih0 (x), K31 = bias
        if (K < HID)                      v0 = Whh0[row * HID + K];
        else if (K >= 16 && K < 19)       v0 = Wih0[row * 3 + (K - 16)];
        else if (K == 31)                 v0 = bih0[row] + bhh0[row];
        // layer1: K0..9 = W_ih1 (h0_new), K16..25 = W_hh1 (h1), K31 = bias
        if (K < HID)                      v1 = Wih1[row * HID + K];
        else if (K >= 16 && K < 16 + HID) v1 = Whh1[row * HID + (K - 16)];
        else if (K == 31)                 v1 = bih1[row] + bhh1[row];
      }
      a0[j] = (_Float16)v0;
      a1[j] = (_Float16)v1;
    }
    A0[g] = a0;
    A1[g] = a1;
  }
  v16h Af1, Af2;
#pragma unroll
  for (int j = 0; j < 16; ++j) {
    const int K = 16 * hi + j;
    float v1 = 0.0f, v2 = 0.0f;
    if (ml < HID) {
      if (K < HID)      v1 = fc1w[ml * HID + K];
      else if (K == 31) v1 = fc1b[ml];
    }
    if (ml < 3) {
      if (K < HID)      v2 = fc2w[ml * HID + K];
      else if (K == 31) v2 = fc2b[ml];
    }
    Af1[j] = (_Float16)v1;
    Af2[j] = (_Float16)v2;
  }

  // ---------------- Phase 3: 18-step 2-layer LSTM over the batch tile --------
  // State in accumulator layout: vgpr r, lanes 0-15 = unit r, lanes 16-31 = unit 8+r.
  float h0[8], h1[8], c0[8], c1[8];
#pragma unroll
  for (int r = 0; r < 8; ++r) { h0[r] = h1[r] = c0[r] = c1[r] = 0.0f; }
  v8f zC;
#pragma unroll
  for (int r = 0; r < 8; ++r) zC[r] = 0.0f;
  const _Float16 Z   = (_Float16)0.0f;
  const _Float16 ONE = (_Float16)1.0f;

  for (int t = 0; t < SEQ; ++t) {
    union { uint2 u; _Float16 h[4]; } q;
    q.u = sFeat[sBase + b16 * SEQ + t];

    // --- B for layer0: lower lanes K0..9 = h0 (units 8,9 via SWAPX16),
    //                   upper lanes K16..18 = x_t, K31 = 1.0 (bias column)
    float p8 = lane_xor16(h0[0]);
    float p9 = lane_xor16(h0[1]);
    v16h B0;
    B0[0] = lo ? (_Float16)h0[0] : q.h[0];
    B0[1] = lo ? (_Float16)h0[1] : q.h[1];
    B0[2] = lo ? (_Float16)h0[2] : q.h[2];
    B0[3] = lo ? (_Float16)h0[3] : Z;
    B0[4] = lo ? (_Float16)h0[4] : Z;
    B0[5] = lo ? (_Float16)h0[5] : Z;
    B0[6] = lo ? (_Float16)h0[6] : Z;
    B0[7] = lo ? (_Float16)h0[7] : Z;
    B0[8] = lo ? (_Float16)p8 : Z;
    B0[9] = lo ? (_Float16)p9 : Z;
    B0[10] = Z; B0[11] = Z; B0[12] = Z; B0[13] = Z; B0[14] = Z;
    B0[15] = lo ? Z : ONE;

    v8f Gi = WMMA_F16(A0[0], B0, zC);
    v8f Gf = WMMA_F16(A0[1], B0, zC);
    v8f Gg = WMMA_F16(A0[2], B0, zC);
    v8f Go = WMMA_F16(A0[3], B0, zC);

    float h0n[8];
#pragma unroll
    for (int r = 0; r < 8; ++r) {
      float cc = sig_(Gf[r]) * c0[r] + sig_(Gi[r]) * tanh_(Gg[r]);
      c0[r] = cc;
      h0n[r] = sig_(Go[r]) * tanh_(cc);
    }

    // --- B for layer1: lower lanes K0..9 = h0_new, upper lanes K16..25 = h1, K31 = 1.0
    float n8 = lane_xor16(h0n[0]);
    float n9 = lane_xor16(h0n[1]);
    float wsh[8];
#pragma unroll
    for (int r = 0; r < 8; ++r) wsh[r] = lane_xor16(h1[r]);
    v16h B1;
#pragma unroll
    for (int r = 0; r < 8; ++r)
      B1[r] = lo ? (_Float16)h0n[r] : (_Float16)wsh[r];
    B1[8]  = lo ? (_Float16)n8 : (_Float16)h1[0];
    B1[9]  = lo ? (_Float16)n9 : (_Float16)h1[1];
    B1[10] = Z; B1[11] = Z; B1[12] = Z; B1[13] = Z; B1[14] = Z;
    B1[15] = lo ? Z : ONE;

    Gi = WMMA_F16(A1[0], B1, zC);
    Gf = WMMA_F16(A1[1], B1, zC);
    Gg = WMMA_F16(A1[2], B1, zC);
    Go = WMMA_F16(A1[3], B1, zC);

#pragma unroll
    for (int r = 0; r < 8; ++r) {
      float cc = sig_(Gf[r]) * c1[r] + sig_(Gi[r]) * tanh_(Gg[r]);
      c1[r] = cc;
      h1[r] = sig_(Go[r]) * tanh_(cc);
      h0[r] = h0n[r];
    }
  }

  // ---------------- Phase 4: FC head (10 -> relu -> 3) via 2 more WMMAs ------
  {
    float p8 = lane_xor16(h1[0]);
    float p9 = lane_xor16(h1[1]);
    v16h Bh;
#pragma unroll
    for (int r = 0; r < 8; ++r) Bh[r] = lo ? (_Float16)h1[r] : Z;
    Bh[8]  = lo ? (_Float16)p8 : Z;
    Bh[9]  = lo ? (_Float16)p9 : Z;
    Bh[10] = Z; Bh[11] = Z; Bh[12] = Z; Bh[13] = Z; Bh[14] = Z;
    Bh[15] = lo ? Z : ONE;

    v8f z = WMMA_F16(Af1, Bh, zC);
#pragma unroll
    for (int r = 0; r < 8; ++r) z[r] = fmaxf(z[r], 0.0f);

    float z8 = lane_xor16(z[0]);
    float z9 = lane_xor16(z[1]);
    v16h Bz;
#pragma unroll
    for (int r = 0; r < 8; ++r) Bz[r] = lo ? (_Float16)z[r] : Z;
    Bz[8]  = lo ? (_Float16)z8 : Z;
    Bz[9]  = lo ? (_Float16)z9 : Z;
    Bz[10] = Z; Bz[11] = Z; Bz[12] = Z; Bz[13] = Z; Bz[14] = Z;
    Bz[15] = lo ? Z : ONE;

    v8f o = WMMA_F16(Af2, Bz, zC);
    if (lo && bG0 < Btot) {
      out[(size_t)bG0 * 3 + 0] = o[0];
      out[(size_t)bG0 * 3 + 1] = o[1];
      out[(size_t)bG0 * 3 + 2] = o[2];
    }
  }
}

extern "C" void kernel_launch(void* const* d_in, const int* in_sizes, int n_in,
                              void* d_out, int out_size, void* d_ws, size_t ws_size,
                              hipStream_t stream) {
  (void)n_in; (void)out_size; (void)d_ws; (void)ws_size;
  const float* x      = (const float*)d_in[0];
  const float* conv_w = (const float*)d_in[1];
  const float* conv_b = (const float*)d_in[2];
  const float* bn_g   = (const float*)d_in[3];
  const float* bn_b   = (const float*)d_in[4];
  const float* bn_m   = (const float*)d_in[5];
  const float* bn_v   = (const float*)d_in[6];
  const float* Wih0   = (const float*)d_in[7];
  const float* Whh0   = (const float*)d_in[8];
  const float* bih0   = (const float*)d_in[9];
  const float* bhh0   = (const float*)d_in[10];
  const float* Wih1   = (const float*)d_in[11];
  const float* Whh1   = (const float*)d_in[12];
  const float* bih1   = (const float*)d_in[13];
  const float* bhh1   = (const float*)d_in[14];
  const float* fc1w   = (const float*)d_in[15];
  const float* fc1b   = (const float*)d_in[16];
  const float* fc2w   = (const float*)d_in[17];
  const float* fc2b   = (const float*)d_in[18];
  float* out = (float*)d_out;

  const int Btot   = in_sizes[0] / 60;           // [B,3,20]
  const int perBlk = WPB * 16;                   // 64 batch rows per block
  const int blocks = (Btot + perBlk - 1) / perBlk;

  hmp_lstm_wmma_kernel<<<blocks, WPB * 32, 0, stream>>>(
      x, conv_w, conv_b, bn_g, bn_b, bn_m, bn_v,
      Wih0, Whh0, bih0, bhh0, Wih1, Whh1, bih1, bhh1,
      fc1w, fc1b, fc2w, fc2b, out, Btot);
}
